// GAT_4123168604187
// MI455X (gfx1250) — matrix operand order repaired and checked by reading, not compiled, not run on hardware
//
#include <hip/hip_runtime.h>
#include <hip/hip_bf16.h>

typedef __attribute__((ext_vector_type(16))) _Float16 v16h;
typedef __attribute__((ext_vector_type(8)))  _Float16 v8h;
typedef __attribute__((ext_vector_type(8)))  float    v8f;

#define NEG_SLOPE 0.2f

// ---- order-preserving float <-> uint key (for atomicMax segment-max) ----
__device__ __forceinline__ unsigned fenc(float f) {
  unsigned u = __float_as_uint(f);
  return (u >> 31) ? ~u : (u | 0x80000000u);
}
__device__ __forceinline__ float fdec(unsigned k) {
  return __uint_as_float((k >> 31) ? (k ^ 0x80000000u) : ~k);
}
// fenc(-inf) == 0x007FFFFF

// =====================================================================
// Operand packing kernels
// =====================================================================
__global__ void f32_to_f16(const float* __restrict__ in, _Float16* __restrict__ out,
                           size_t n) {
  size_t t = (size_t)blockIdx.x * blockDim.x + threadIdx.x;
  if (t < n) out[t] = (_Float16)in[t];
}

// W[K,Nn] f32 row-major -> Wt[Nn,K] f16
__global__ void transpose_to_f16(const float* __restrict__ W, _Float16* __restrict__ Wt,
                                 int K, int Nn) {
  int t = blockIdx.x * blockDim.x + threadIdx.x;
  if (t >= K * Nn) return;
  int k = t / Nn, n = t - k * Nn;
  Wt[(size_t)n * K + k] = (_Float16)W[t];
}

// =====================================================================
// Dense GEMM: C[M,Nn] = A[M,K] x B[K,Nn] with Ah f16 row-major and
// Bt f16 = B transposed ([Nn,K] row-major). f32 accumulate via
// v_wmma_f32_16x16x32_f16. One wave computes a 16x64 tile.
// Requires M%16==0, Nn%64==0, K%32==0.
// =====================================================================
__global__ void gemm_wmma_f16(const _Float16* __restrict__ Ah,
                              const _Float16* __restrict__ Bt,
                              float* __restrict__ C,
                              int M, int Nn, int K) {
  const int wave  = (blockIdx.x * blockDim.x + threadIdx.x) >> 5;
  const int lane  = threadIdx.x & 31;
  const int ncol  = Nn >> 6;                // number of 64-wide column tiles
  const int mTile = wave / ncol;
  const int nTile = wave - mTile * ncol;
  if (mTile * 16 >= M) return;
  const int m0 = mTile * 16;
  const int n0 = nTile * 64;
  const int g  = lane >> 4;                 // half-wave group (0/1)
  const int r  = lane & 15;                 // M-row (A) / N-col (B, C/D)

  // A fragment layout: halfs[j] = K = k0+16g+j (contiguous).
  const _Float16* arow = Ah + (size_t)(m0 + r) * K + 16 * g;
  // B fragment layout: half[j] = B[k0+16g+j, n] = Bt[n, k0+16g+j] (contiguous).
  const _Float16* b0 = Bt + (size_t)(n0 + r) * K + 16 * g;

  v8f acc[4] = {v8f{}, v8f{}, v8f{}, v8f{}};

  for (int k0 = 0; k0 < K; k0 += 32) {
    v8h a0 = *(const v8h*)(arow + k0);
    v8h a1 = *(const v8h*)(arow + k0 + 8);
    v16h a = __builtin_shufflevector(a0, a1, 0, 1, 2, 3, 4, 5, 6, 7,
                                     8, 9, 10, 11, 12, 13, 14, 15);
    __builtin_prefetch(arow + k0 + 32, 0, 0);   // next K-step A chunk -> L2/L0
#pragma unroll
    for (int t = 0; t < 4; ++t) {
      v16h b = *(const v16h*)(b0 + (size_t)(16 * t) * K + k0);
      acc[t] = __builtin_amdgcn_wmma_f32_16x16x32_f16(
          false, a, false, b, (short)0, acc[t], false, false);
    }
  }
  // C/D layout: lane(g, n=r), VGPR i -> row m0 + i + 8g
#pragma unroll
  for (int t = 0; t < 4; ++t)
#pragma unroll
    for (int i = 0; i < 8; ++i)
      C[(size_t)(m0 + i + 8 * g) * Nn + n0 + 16 * t + r] = acc[t][i];
}

// =====================================================================
// el[n,h] = sum_d feat[n,h,d]*al[h,d] ; er likewise. One thread per (n,h).
// =====================================================================
__global__ void attn_scores(const float* __restrict__ feat,
                            const float* __restrict__ al,
                            const float* __restrict__ ar,
                            float* __restrict__ el, float* __restrict__ er,
                            int Nn, int H, int D) {
  int t = blockIdx.x * blockDim.x + threadIdx.x;
  if (t >= Nn * H) return;
  int n = t / H, h = t - n * H;
  const float* f = feat + (size_t)n * H * D + (size_t)h * D;
  const float* L = al + (size_t)h * D;
  const float* R = ar + (size_t)h * D;
  float sl = 0.f, sr = 0.f;
  for (int d = 0; d < D; ++d) { float v = f[d]; sl += v * L[d]; sr += v * R[d]; }
  el[t] = sl;
  er[t] = sr;
}

// =====================================================================
// Edge pass 1: segment-max of leaky_relu(el[src]+er[dst]) into m[dst,h].
// =====================================================================
__global__ void edge_max(const int* __restrict__ src, const int* __restrict__ dst,
                         const float* __restrict__ el, const float* __restrict__ er,
                         unsigned* __restrict__ m, int E_, int H) {
  int t = blockIdx.x * blockDim.x + threadIdx.x;
  if (t >= E_ * H) return;
  int e = t / H, h = t - e * H;
  float v = el[(size_t)src[e] * H + h] + er[(size_t)dst[e] * H + h];
  v = v > 0.f ? v : NEG_SLOPE * v;
  atomicMax(&m[(size_t)dst[e] * H + h], fenc(v));
}

// =====================================================================
// Edge pass 2: ex = exp(e - max[dst]); store per-edge, atomic segment-sum.
// =====================================================================
__global__ void edge_exp(const int* __restrict__ src, const int* __restrict__ dst,
                         const float* __restrict__ el, const float* __restrict__ er,
                         const unsigned* __restrict__ m, float* __restrict__ s,
                         float* __restrict__ ex, int E_, int H) {
  int t = blockIdx.x * blockDim.x + threadIdx.x;
  if (t >= E_ * H) return;
  int e = t / H, h = t - e * H;
  int dn = dst[e];
  float v = el[(size_t)src[e] * H + h] + er[(size_t)dn * H + h];
  v = v > 0.f ? v : NEG_SLOPE * v;
  float x = expf(v - fdec(m[(size_t)dn * H + h]));
  ex[t] = x;
  atomicAdd(&s[(size_t)dn * H + h], x);
}

// =====================================================================
// Aggregation: agg[dst] += feat[src] * (ex/s[dst]), one wave per edge.
// Each lane owns a contiguous HD/32 chunk of the feature row.
// =====================================================================
template <int HD, int D>
__global__ void aggregate(const int* __restrict__ src, const int* __restrict__ dst,
                          const float* __restrict__ feat,
                          const float* __restrict__ ex, const float* __restrict__ s,
                          float* __restrict__ agg, int E_) {
  constexpr int CH = HD / 32;
  constexpr int H  = HD / D;
  int wave = (blockIdx.x * blockDim.x + threadIdx.x) >> 5;
  int lane = threadIdx.x & 31;
  if (wave >= E_) return;
  int sn = src[wave], dn = dst[wave];
  int f0 = lane * CH;
  int h  = f0 / D;                         // chunk lies entirely inside one head
  float a = ex[(size_t)wave * H + h] / s[(size_t)dn * H + h];
  const float* fs = feat + (size_t)sn * HD + f0;
  float* ag = agg + (size_t)dn * HD + f0;
#pragma unroll
  for (int i = 0; i < CH; ++i) atomicAdd(&ag[i], fs[i] * a);
}

// =====================================================================
// Epilogues and fills
// =====================================================================
// out_h = (f16) elu(buf + bias) : feeds the next GEMM directly.
__global__ void bias_elu_to_f16(const float* __restrict__ buf,
                                const float* __restrict__ bias,
                                _Float16* __restrict__ out_h,
                                size_t total, int F) {
  size_t t = (size_t)blockIdx.x * blockDim.x + threadIdx.x;
  if (t >= total) return;
  int f = (int)(t % F);
  float v = buf[t] + bias[f];
  out_h[t] = (_Float16)(v > 0.f ? v : expm1f(v));   // jax.nn.elu, alpha=1
}

__global__ void bias_out(const float* __restrict__ agg, const float* __restrict__ bias,
                         float* __restrict__ out, size_t total, int F) {
  size_t t = (size_t)blockIdx.x * blockDim.x + threadIdx.x;
  if (t >= total) return;
  int f = (int)(t % F);
  out[t] = agg[t] + bias[f];
}

__global__ void fill_u32(unsigned* __restrict__ p, unsigned v, size_t n) {
  size_t t = (size_t)blockIdx.x * blockDim.x + threadIdx.x;
  if (t < n) p[t] = v;
}
__global__ void fill_f32(float* __restrict__ p, float v, size_t n) {
  size_t t = (size_t)blockIdx.x * blockDim.x + threadIdx.x;
  if (t < n) p[t] = v;
}

// =====================================================================
extern "C" void kernel_launch(void* const* d_in, const int* in_sizes, int n_in,
                              void* d_out, int out_size, void* d_ws, size_t ws_size,
                              hipStream_t stream) {
  (void)n_in; (void)out_size; (void)ws_size;
  const float* x   = (const float*)d_in[0];
  const int*   src = (const int*)d_in[1];
  const int*   dst = (const int*)d_in[2];
  const float* W1  = (const float*)d_in[3];
  const float* al1 = (const float*)d_in[4];
  const float* ar1 = (const float*)d_in[5];
  const float* b1  = (const float*)d_in[6];
  const float* W2  = (const float*)d_in[7];
  const float* al2 = (const float*)d_in[8];
  const float* ar2 = (const float*)d_in[9];
  const float* b2  = (const float*)d_in[10];
  float* out = (float*)d_out;

  const int IN = 512, H1 = 4, D1 = 64, HD1 = 256, H2 = 1, D2 = 64, HD2 = 64;
  const int N = in_sizes[0] / IN;   // 50000
  const int E = in_sizes[1];        // 800000

  // ---- workspace carve-out (256B aligned) ----
  char* w = (char*)d_ws;
  auto take = [&](size_t bytes) -> char* {
    char* p = w; w += (bytes + 255) & ~(size_t)255; return p;
  };
  float*    feat1 = (float*)   take((size_t)N * HD1 * 4);
  float*    h1    = (float*)   take((size_t)N * HD1 * 4);  // agg1 (f32)
  float*    feat2 = (float*)   take((size_t)N * HD2 * 4);
  float*    agg2  = (float*)   take((size_t)N * HD2 * 4);
  float*    el1   = (float*)   take((size_t)N * H1 * 4);
  float*    er1   = (float*)   take((size_t)N * H1 * 4);
  unsigned* m1    = (unsigned*)take((size_t)N * H1 * 4);
  float*    s1    = (float*)   take((size_t)N * H1 * 4);
  float*    ex1   = (float*)   take((size_t)E * H1 * 4);
  float*    el2   = (float*)   take((size_t)N * H2 * 4);
  float*    er2   = (float*)   take((size_t)N * H2 * 4);
  unsigned* m2    = (unsigned*)take((size_t)N * H2 * 4);
  float*    s2    = (float*)   take((size_t)N * H2 * 4);
  float*    ex2   = (float*)   take((size_t)E * H2 * 4);
  _Float16* W1t   = (_Float16*)take((size_t)IN * HD1 * 2);   // [HD1, IN]
  _Float16* W2t   = (_Float16*)take((size_t)HD1 * HD2 * 2);  // [HD2, HD1]
  // Aliases (stream-ordered lifetimes, no overlap in time):
  //  xh  lives only during GEMM1; h1 is first written (zero-filled) after GEMM1.
  //  h1h lives from bias_elu onward; feat1 is dead after aggregate<HD1>.
  _Float16* xh  = (_Float16*)h1;     // N*IN*2  == N*HD1*4 bytes
  _Float16* h1h = (_Float16*)feat1;  // N*HD1*2 <= N*HD1*4 bytes

  const int BLK = 256;
  auto blocks = [&](size_t work) -> unsigned { return (unsigned)((work + BLK - 1) / BLK); };
  const unsigned NEG_INF_KEY = 0x007FFFFFu;  // fenc(-inf)

  // ---- pack operands to f16 ----
  f32_to_f16<<<blocks((size_t)N * IN), BLK, 0, stream>>>(x, xh, (size_t)N * IN);
  transpose_to_f16<<<blocks((size_t)IN * HD1), BLK, 0, stream>>>(W1, W1t, IN, HD1);
  transpose_to_f16<<<blocks((size_t)HD1 * HD2), BLK, 0, stream>>>(W2, W2t, HD1, HD2);

  // ================= Layer 1 =================
  {
    int waves = (N / 16) * (HD1 / 64);
    gemm_wmma_f16<<<blocks((size_t)waves * 32), BLK, 0, stream>>>(xh, W1t, feat1, N, HD1, IN);
  }
  attn_scores<<<blocks((size_t)N * H1), BLK, 0, stream>>>(feat1, al1, ar1, el1, er1, N, H1, D1);
  fill_u32<<<blocks((size_t)N * H1), BLK, 0, stream>>>(m1, NEG_INF_KEY, (size_t)N * H1);
  fill_f32<<<blocks((size_t)N * H1), BLK, 0, stream>>>(s1, 0.f, (size_t)N * H1);
  fill_f32<<<blocks((size_t)N * HD1), BLK, 0, stream>>>(h1, 0.f, (size_t)N * HD1);  // xh dead
  edge_max<<<blocks((size_t)E * H1), BLK, 0, stream>>>(src, dst, el1, er1, m1, E, H1);
  edge_exp<<<blocks((size_t)E * H1), BLK, 0, stream>>>(src, dst, el1, er1, m1, s1, ex1, E, H1);
  aggregate<HD1, D1><<<blocks((size_t)E * 32), BLK, 0, stream>>>(src, dst, feat1, ex1, s1, h1, E);
  // feat1 dead from here; write f16 activations into its space.
  bias_elu_to_f16<<<blocks((size_t)N * HD1), BLK, 0, stream>>>(h1, b1, h1h, (size_t)N * HD1, HD1);

  // ================= Layer 2 =================
  {
    int waves = (N / 16) * (HD2 / 64);
    gemm_wmma_f16<<<blocks((size_t)waves * 32), BLK, 0, stream>>>(h1h, W2t, feat2, N, HD2, HD1);
  }
  attn_scores<<<blocks((size_t)N * H2), BLK, 0, stream>>>(feat2, al2, ar2, el2, er2, N, H2, D2);
  fill_u32<<<blocks((size_t)N * H2), BLK, 0, stream>>>(m2, NEG_INF_KEY, (size_t)N * H2);
  fill_f32<<<blocks((size_t)N * H2), BLK, 0, stream>>>(s2, 0.f, (size_t)N * H2);
  fill_f32<<<blocks((size_t)N * HD2), BLK, 0, stream>>>(agg2, 0.f, (size_t)N * HD2);
  edge_max<<<blocks((size_t)E * H2), BLK, 0, stream>>>(src, dst, el2, er2, m2, E, H2);
  edge_exp<<<blocks((size_t)E * H2), BLK, 0, stream>>>(src, dst, el2, er2, m2, s2, ex2, E, H2);
  aggregate<HD2, D2><<<blocks((size_t)E * 32), BLK, 0, stream>>>(src, dst, feat2, ex2, s2, agg2, E);
  // mean over a single head is identity: out = agg2 + b2
  bias_out<<<blocks((size_t)N * HD2), BLK, 0, stream>>>(agg2, b2, out, (size_t)N * HD2, HD2);
}
